// GAT_SS_86887188398714
// MI455X (gfx1250) — matrix operand order repaired and checked by reading, not compiled, not run on hardware
//
#include <hip/hip_runtime.h>
#include <hip/hip_bf16.h>
#include <math.h>

// ---------------------------------------------------------------------------
// GAT + self-supervised structural branch for MI455X (gfx1250, wave32).
// GEMMs run on v_wmma_f32_16x16x32_f16 (f16 in / f32 acc) with N-register
// blocking (one A fragment feeds NBLK B tiles -> 4x less A traffic from L2).
// Edge aggregation is fused exp(-leakyrelu)-weighted segment-sum with f32
// global atomics that stay resident in the 192MB L2 (h is only 102MB).
// ---------------------------------------------------------------------------

typedef __attribute__((ext_vector_type(16))) _Float16 v16h;
typedef __attribute__((ext_vector_type(8)))  _Float16 v8h;
typedef __attribute__((ext_vector_type(8)))  float    v8f;

#define NNODES 50000
#define NEDGES 1600000
#define NFEAT  512
#define NHID   64
#define NCLASS 16
#define NSTRUC 64
#define NHEADS 8
#define NC1    (NHEADS * NHID)     // 512 (concat width)
#define NC2    (NCLASS + NSTRUC)   // 80  ([W_out | W_enc] fused GEMM)
#define ALPHA  0.2f
#define NIDX   5000

__device__ __forceinline__ float lrelu_neg_exp(float s) {
    float l = s > 0.0f ? s : ALPHA * s;
    return __expf(-l);
}
__device__ __forceinline__ float elu1(float v) {
    return v > 0.0f ? v : (__expf(v) - 1.0f);
}

// --------------------------- f32 -> f16 cast --------------------------------
__global__ void k_cast_f16(const float* __restrict__ src, _Float16* __restrict__ dst, long n) {
    long i = (long)blockIdx.x * blockDim.x + threadIdx.x;
    if (i < n) dst[i] = (_Float16)src[i];
}

// ----------------- pack weights into WMMA B-fragment layout -----------------
// Packed element t = ((tn*KS + ks)*32 + lane)*16 + e  holds
//   B_logical[K = ks*32 + e + (lane>>4)*16][Ncol = tn*16 + (lane&15)]
__global__ void k_pack_w1(const float* __restrict__ Wh /*[8,512,64]*/,
                          _Float16* __restrict__ bp, int total) {
    int t = blockIdx.x * blockDim.x + threadIdx.x;
    if (t >= total) return;
    int e    = t & 15;
    int lane = (t >> 4) & 31;
    int ks   = (t >> 9) & 15;
    int tn   = t >> 13;
    int ncol = tn * 16 + (lane & 15);
    int kk   = ks * 32 + e + ((lane >> 4) << 4);
    int head = ncol >> 6, j = ncol & 63;
    bp[t] = (_Float16)Wh[(size_t)head * NFEAT * NHID + (size_t)kk * NHID + j];
}

__global__ void k_pack_w2(const float* __restrict__ Wout /*[512,16]*/,
                          const float* __restrict__ Wenc /*[512,64]*/,
                          _Float16* __restrict__ bp, int total) {
    int t = blockIdx.x * blockDim.x + threadIdx.x;
    if (t >= total) return;
    int e    = t & 15;
    int lane = (t >> 4) & 31;
    int ks   = (t >> 9) & 15;
    int tn   = t >> 13;
    int ncol = tn * 16 + (lane & 15);
    int kk   = ks * 32 + e + ((lane >> 4) << 4);
    float v = (ncol < NCLASS) ? Wout[kk * NCLASS + ncol]
                              : Wenc[kk * NSTRUC + (ncol - NCLASS)];
    bp[t] = (_Float16)v;
}

// ---------------------- WMMA GEMM: C[M,NC] = A[M,K] x B ---------------------
// One wave per 16x(16*NBLK) output strip: one A fragment feeds NBLK B tiles,
// K swept in steps of 32 via v_wmma_f32_16x16x32_f16. A row-major f16,
// B pre-packed into per-lane-contiguous fragments (two b128 loads each).
template <int NBLK>
__global__ void k_wmma_gemm(const _Float16* __restrict__ A,
                            const _Float16* __restrict__ Bp,
                            float* __restrict__ C,
                            int M, int K, int NC) {
    const int lane = threadIdx.x & 31;
    const int wave = blockIdx.x * (blockDim.x >> 5) + (threadIdx.x >> 5);
    const int ntN  = NC >> 4;           // n-tiles total (divisible by NBLK)
    const int ngrp = ntN / NBLK;        // n-tile groups
    const int strips = (M >> 4) * ngrp;
    if (wave >= strips) return;         // whole-wave uniform: EXEC stays all-1s
    const int tm  = wave / ngrp;
    const int tn0 = (wave % ngrp) * NBLK;
    const int g   = lane >> 4;
    const int mrow = tm * 16 + (lane & 15);
    const int ksteps = K >> 5;
    const _Float16* __restrict__ arow = A + (size_t)mrow * K;

    v8f acc[NBLK];
#pragma unroll
    for (int b = 0; b < NBLK; ++b) acc[b] = v8f{};

    for (int ks = 0; ks < ksteps; ++ks) {
        const int k0 = ks << 5;
        // A frag, contiguous k order (same as B pack): lane(g,m) holds K = {k0+16g..+15}
        v8h a0 = *(const v8h*)(arow + k0 + g * 16);
        v8h a1 = *(const v8h*)(arow + k0 + g * 16 + 8);
        v16h afrag;
#pragma unroll
        for (int i = 0; i < 8; ++i) { afrag[i] = a0[i]; afrag[8 + i] = a1[i]; }
#pragma unroll
        for (int b = 0; b < NBLK; ++b) {
            const v8h* bptr = (const v8h*)(Bp +
                ((size_t)(tn0 + b) * ksteps + ks) * 512 + lane * 16);
            v8h b0 = bptr[0], b1 = bptr[1];
            v16h bfrag;
#pragma unroll
            for (int i = 0; i < 8; ++i) { bfrag[i] = b0[i]; bfrag[8 + i] = b1[i]; }
            acc[b] = __builtin_amdgcn_wmma_f32_16x16x32_f16(
                         false, afrag, false, bfrag, (short)0, acc[b], false, false);
        }
    }
    // C/D f32 layout: VGPR r -> M = g*8 + r, N = lane&15
    const int mbase = tm * 16 + g * 8;
    const int nlow  = lane & 15;
#pragma unroll
    for (int b = 0; b < NBLK; ++b) {
        const int nout = (tn0 + b) * 16 + nlow;
#pragma unroll
        for (int r = 0; r < 8; ++r)
            C[(size_t)(mbase + r) * NC + nout] = acc[b][r];
    }
}

// ----------------- layer-1 attention logits per (head, node) ----------------
__global__ void k_scores1(const float* __restrict__ hw, const float* __restrict__ a_heads,
                          float* __restrict__ ssrc, float* __restrict__ sdst) {
    int t = blockIdx.x * blockDim.x + threadIdx.x;
    if (t >= NHEADS * NNODES) return;
    int h = t / NNODES, n = t % NNODES;
    const float* hv = hw + (size_t)n * NC1 + h * NHID;
    const float* av = a_heads + h * 2 * NHID;
    float s1 = 0.f, s2 = 0.f;
#pragma unroll 8
    for (int j = 0; j < NHID; ++j) { s1 += hv[j] * av[j]; s2 += hv[j] * av[NHID + j]; }
    ssrc[t] = s1; sdst[t] = s2;
}

// ------------- layer-1 fused edge kernel: all 8 heads, 64 feats -------------
__global__ void k_edge1(const int* __restrict__ src, const int* __restrict__ dst,
                        const float* __restrict__ ssrc, const float* __restrict__ sdst,
                        const float* __restrict__ hw,
                        float* __restrict__ rowsum, float* __restrict__ hp) {
    long t = (long)blockIdx.x * blockDim.x + threadIdx.x;
    if (t >= (long)NEDGES * NHID) return;
    int e = (int)(t >> 6);
    int j = (int)(t & 63);
    int s = src[e], d = dst[e];
#pragma unroll
    for (int h = 0; h < NHEADS; ++h) {
        float w = lrelu_neg_exp(ssrc[h * NNODES + s] + sdst[h * NNODES + d]);
        if (j == 0) atomicAdd(&rowsum[h * NNODES + s], w);
        atomicAdd(&hp[(size_t)s * NC1 + h * NHID + j],
                  w * hw[(size_t)d * NC1 + h * NHID + j]);
    }
}

// -------- layer-1 epilogue: rownorm + ELU, emit f16 for next GEMM -----------
__global__ void k_norm_elu1(const float* __restrict__ hp, const float* __restrict__ rowsum,
                            _Float16* __restrict__ hcat) {
    long t = (long)blockIdx.x * blockDim.x + threadIdx.x;
    if (t >= (long)NNODES * NC1) return;
    int n = (int)(t / NC1), c = (int)(t % NC1);
    float rs = rowsum[(c >> 6) * NNODES + n] + 1e-16f;
    hcat[t] = (_Float16)elu1(hp[t] / rs);
}

// --------- per-node logits for layer-2 heads + structural projections -------
__global__ void k_scores2(const float* __restrict__ out2 /*[N,80]*/,
                          const float* __restrict__ a_out, const float* __restrict__ b_enc,
                          const float* __restrict__ W_ss, const float* __restrict__ struc,
                          float* __restrict__ s2s, float* __restrict__ s2d,
                          float* __restrict__ ty, float* __restrict__ tsf) {
    int n = blockIdx.x * blockDim.x + threadIdx.x;
    if (n >= NNODES) return;
    const float* o = out2 + (size_t)n * NC2;
    float s1 = 0.f, s2 = 0.f;
#pragma unroll
    for (int c = 0; c < NCLASS; ++c) { s1 += o[c] * a_out[c]; s2 += o[c] * a_out[NCLASS + c]; }
    float tv = 0.f, sv = 0.f;
#pragma unroll 8
    for (int jj = 0; jj < NSTRUC; ++jj) {
        tv += (o[NCLASS + jj] + b_enc[jj]) * W_ss[jj];
        sv += struc[(size_t)n * NSTRUC + jj] * W_ss[jj];
    }
    s2s[n] = s1; s2d[n] = s2; ty[n] = tv; tsf[n] = sv;
}

// -------- layer-2 fused edge kernel: class head + y-branch + sf-branch ------
__global__ void k_edge2(const int* __restrict__ src, const int* __restrict__ dst,
                        const float* __restrict__ s2s, const float* __restrict__ s2d,
                        const float* __restrict__ out2,
                        const float* __restrict__ ty, const float* __restrict__ tsf,
                        const float* __restrict__ a_ss,
                        float* __restrict__ rs2, float* __restrict__ hp2,
                        float* __restrict__ rsy, float* __restrict__ agy,
                        float* __restrict__ rssf, float* __restrict__ agsf) {
    int e = blockIdx.x * blockDim.x + threadIdx.x;
    if (e >= NEDGES) return;
    int s = src[e], d = dst[e];
    float a0 = a_ss[0], a1 = a_ss[1];
    // classification head
    float w = lrelu_neg_exp(s2s[s] + s2d[d]);
    atomicAdd(&rs2[s], w);
    const float* hd = out2 + (size_t)d * NC2;
#pragma unroll
    for (int c = 0; c < NCLASS; ++c) atomicAdd(&hp2[(size_t)s * NCLASS + c], w * hd[c]);
    // structural y branch (d=1)
    float wy = lrelu_neg_exp(ty[s] * a0 + ty[d] * a1);
    atomicAdd(&rsy[s], wy); atomicAdd(&agy[s], wy * ty[d]);
    // structural sf branch (d=1)
    float wf = lrelu_neg_exp(tsf[s] * a0 + tsf[d] * a1);
    atomicAdd(&rssf[s], wf); atomicAdd(&agsf[s], wf * tsf[d]);
}

// ------------- finalize: xo to d_out, z accumulated into d_out[1] -----------
__global__ void k_finalize(const float* __restrict__ hp2, const float* __restrict__ rs2,
                           const float* __restrict__ agy, const float* __restrict__ rsy,
                           const float* __restrict__ agsf, const float* __restrict__ rssf,
                           float* __restrict__ dout) {
    int n = blockIdx.x * blockDim.x + threadIdx.x;
    if (n >= NNODES) return;
    float rs = rs2[n] + 1e-16f;
#pragma unroll
    for (int c = 0; c < NCLASS; ++c)
        dout[2 + (size_t)n * NCLASS + c] = elu1(hp2[(size_t)n * NCLASS + c] / rs);
    float yv  = elu1(agy[n]  / (rsy[n]  + 1e-16f));
    float sfv = elu1(agsf[n] / (rssf[n] + 1e-16f));
    float dz = yv - sfv;
    atomicAdd(&dout[1], dz * dz * (1.0f / (float)NNODES));
}

// --------------------- NLL over sampled idx (log_softmax) -------------------
__global__ void k_nll(const int* __restrict__ idx, const int* __restrict__ labels,
                      float* __restrict__ dout) {
    int i = blockIdx.x * blockDim.x + threadIdx.x;
    if (i >= NIDX) return;
    int row = idx[i];
    int lab = labels[row];
    const float* xr = dout + 2 + (size_t)row * NCLASS;
    float m = xr[0];
#pragma unroll
    for (int c = 1; c < NCLASS; ++c) m = fmaxf(m, xr[c]);
    float ssum = 0.f;
#pragma unroll
    for (int c = 0; c < NCLASS; ++c) ssum += __expf(xr[c] - m);
    float lp = xr[lab] - (__logf(ssum) + m);
    atomicAdd(&dout[0], -lp * (1.0f / (float)NIDX));
}

// ---------------------------------------------------------------------------
extern "C" void kernel_launch(void* const* d_in, const int* in_sizes, int n_in,
                              void* d_out, int out_size, void* d_ws, size_t ws_size,
                              hipStream_t stream) {
    const float* x        = (const float*)d_in[0];
    const int*   eidx     = (const int*)d_in[1];
    const float* struc    = (const float*)d_in[2];
    const int*   idx      = (const int*)d_in[3];
    const int*   labels   = (const int*)d_in[4];
    const float* W_heads  = (const float*)d_in[5];
    const float* a_heads  = (const float*)d_in[6];
    const float* W_out    = (const float*)d_in[7];
    const float* a_out    = (const float*)d_in[8];
    const float* W_enc    = (const float*)d_in[9];
    const float* b_enc    = (const float*)d_in[10];
    const float* W_ss     = (const float*)d_in[11];
    const float* a_ss     = (const float*)d_in[12];
    float* dout = (float*)d_out;
    const int* src = eidx;
    const int* dst = eidx + NEDGES;

    // ---- workspace bump allocator (256B aligned) ----
    char* base = (char*)d_ws;
    size_t off = 0;
    auto alloc = [&](size_t bytes) -> char* {
        char* p = base + off;
        off = (off + bytes + 255) & ~(size_t)255;
        return p;
    };
    _Float16* xh   = (_Float16*)alloc((size_t)NNODES * NFEAT * 2); // reused as hcat_f16
    _Float16* b1p  = (_Float16*)alloc((size_t)NFEAT * NC1 * 2);
    _Float16* b2p  = (_Float16*)alloc((size_t)NFEAT * NC2 * 2);
    float* hw      = (float*)alloc((size_t)NNODES * NC1 * 4);
    float* hp      = (float*)alloc((size_t)NNODES * NC1 * 4);     // reused as out2[N,80]
    float* ssrc    = (float*)alloc((size_t)NHEADS * NNODES * 4);
    float* sdst    = (float*)alloc((size_t)NHEADS * NNODES * 4);
    float* rowsum1 = (float*)alloc((size_t)NHEADS * NNODES * 4);
    float* s2s     = (float*)alloc((size_t)NNODES * 4);
    float* s2d     = (float*)alloc((size_t)NNODES * 4);
    float* ty      = (float*)alloc((size_t)NNODES * 4);
    float* tsf     = (float*)alloc((size_t)NNODES * 4);
    // zeroed as one contiguous block: rs2, rsy, agy, rssf, agsf, hp2
    float* zblock  = (float*)alloc((size_t)(5 + NCLASS) * NNODES * 4);
    float* rs2  = zblock;
    float* rsy  = zblock + (size_t)NNODES;
    float* agy  = zblock + (size_t)2 * NNODES;
    float* rssf = zblock + (size_t)3 * NNODES;
    float* agsf = zblock + (size_t)4 * NNODES;
    float* hp2  = zblock + (size_t)5 * NNODES;
    float* out2 = hp;           // gemm2 output reuses hp region (hp dead by then)
    _Float16* hcat = xh;        // layer-1 output reuses xh region (xh dead by then)

    const int TB = 256;
    auto blocks = [](long n, int tb) { return (int)((n + tb - 1) / tb); };

    // 1) cast x -> f16
    k_cast_f16<<<blocks((long)NNODES * NFEAT, TB), TB, 0, stream>>>(x, xh, (long)NNODES * NFEAT);
    // 2) pack weights into WMMA B-fragment layouts
    k_pack_w1<<<blocks((long)NFEAT * NC1, TB), TB, 0, stream>>>(W_heads, b1p, NFEAT * NC1);
    k_pack_w2<<<blocks((long)NFEAT * NC2, TB), TB, 0, stream>>>(W_out, W_enc, b2p, NFEAT * NC2);
    // 3) GEMM 1: hw[N,512] = x @ [W_head0 | ... | W_head7]; 16x64 strips (NBLK=4)
    {
        int strips = (NNODES / 16) * ((NC1 / 16) / 4);
        k_wmma_gemm<4><<<blocks((long)strips, 8), TB, 0, stream>>>(xh, b1p, hw,
                                                                   NNODES, NFEAT, NC1);
    }
    // 4) per-(head,node) attention logits
    k_scores1<<<blocks((long)NHEADS * NNODES, TB), TB, 0, stream>>>(hw, a_heads, ssrc, sdst);
    // 5) zero accumulators, fused edge aggregation (8 heads x 64 feats)
    hipMemsetAsync(rowsum1, 0, (size_t)NHEADS * NNODES * 4, stream);
    hipMemsetAsync(hp, 0, (size_t)NNODES * NC1 * 4, stream);
    k_edge1<<<blocks((long)NEDGES * NHID, TB), TB, 0, stream>>>(src, dst, ssrc, sdst, hw, rowsum1, hp);
    // 6) row-normalize + ELU -> f16 concat features
    k_norm_elu1<<<blocks((long)NNODES * NC1, TB), TB, 0, stream>>>(hp, rowsum1, hcat);
    // 7) GEMM 2: out2[N,80] = hcat @ [W_out | W_enc]; 16x80 strips (NBLK=5)
    {
        int strips = (NNODES / 16) * ((NC2 / 16) / 5);
        k_wmma_gemm<5><<<blocks((long)strips, 8), TB, 0, stream>>>(hcat, b2p, out2,
                                                                   NNODES, NFEAT, NC2);
    }
    // 8) per-node layer-2 logits + structural projections
    k_scores2<<<blocks((long)NNODES, TB), TB, 0, stream>>>(out2, a_out, b_enc, W_ss, struc,
                                                           s2s, s2d, ty, tsf);
    // 9) zero layer-2 accumulators, fused edge kernel for all three branches
    hipMemsetAsync(zblock, 0, (size_t)(5 + NCLASS) * NNODES * 4, stream);
    k_edge2<<<blocks((long)NEDGES, TB), TB, 0, stream>>>(src, dst, s2s, s2d, out2, ty, tsf,
                                                         a_ss, rs2, hp2, rsy, agy, rssf, agsf);
    // 10) finalize outputs: d_out = [nll, z, xo(N x 16)]
    hipMemsetAsync(dout, 0, 2 * sizeof(float), stream);
    k_finalize<<<blocks((long)NNODES, TB), TB, 0, stream>>>(hp2, rs2, agy, rsy, agsf, rssf, dout);
    k_nll<<<blocks((long)NIDX, TB), TB, 0, stream>>>(idx, labels, dout);
}
